// Transformer_44418551775388
// MI455X (gfx1250) — matrix operand order repaired and checked by reading, not compiled, not run on hardware
//
#include <hip/hip_runtime.h>

// ---------------------------------------------------------------------------
// Transformer forward for MI455X (gfx1250, wave32, WMMA, async global->LDS).
//
// Design for MI455X:
//  * ~233 GFLOP of GEMM work, ~150 MB resident data (fits 192 MB L2) ->
//    compute-bound on the matrix pipes. All GEMMs use v_wmma_f32_16x16x32_f16.
//  * All weights/activations are converted to f16 ONCE per launch (a ~150 MB
//    streaming pass, ~10us at 23.3 TB/s), so GEMM tiles are pure f16 and are
//    staged with GLOBAL_LOAD_ASYNC_TO_LDS_B128 (ASYNCcnt + s_wait_asynccnt).
//  * Double-buffered LDS + software-pipelined async copies: tile k+1 is in
//    flight (ASYNCcnt > 0) while tile k runs through the WMMAs; since async
//    loads complete in order, s_wait_asynccnt <= ops-per-tile is exactly the
//    "previous tile has landed" condition.
//  * Workspace: ~200 MB assumed available in d_ws (f16 weights 121 MB +
//    activations + f32 scratch).
// ---------------------------------------------------------------------------

typedef __attribute__((ext_vector_type(16))) _Float16 v16h;
typedef __attribute__((ext_vector_type(8)))  _Float16 v8h;
typedef __attribute__((ext_vector_type(8)))  float    v8f;

#define TILE_M 128
#define TILE_N 128
#define TILE_K 32

// ---- CDNA5 async global->LDS copy (16B per lane), tracked by ASYNCcnt ----
__device__ __forceinline__ void async_copy16(unsigned lds_off, const void* gaddr)
{
    unsigned long long ga = (unsigned long long)(size_t)gaddr;
    asm volatile("global_load_async_to_lds_b128 %0, %1, off"
                 :: "v"(lds_off), "v"(ga)
                 : "memory");
}

#if __has_builtin(__builtin_amdgcn_s_wait_asynccnt)
#define WAIT_ASYNC(n) __builtin_amdgcn_s_wait_asynccnt(n)
#else
#define WAIT_ASYNC(n) asm volatile("s_wait_asynccnt %0" :: "i"(n) : "memory")
#endif

// ---------------------------------------------------------------------------
// C[M,N] = alpha * A[M,K] x B + bias,  B is (N,K) if BT else (K,N). All f16
// inputs, f32 accumulate; epilogue writes f32 (Cf) and/or f16 (Ch).
// Batched: grid.z decomposed as (b = z/nh, h = z%nh) with separate strides.
// ---------------------------------------------------------------------------
template<bool BT, bool RELU>
__global__ void __launch_bounds__(256)
gemm_wmma_kernel(const _Float16* __restrict__ Abase, int lda, long sAb, long sAh,
                 const _Float16* __restrict__ Bbase, int ldb, long sBb, long sBh,
                 const float* __restrict__ bias,
                 float* __restrict__ Cfbase, _Float16* __restrict__ Chbase,
                 int ldc, long sCb, long sCh,
                 int M, int N, int K, float alpha, int nh)
{
    // double-buffered; rows are 64B (32 halves) so every 16B chunk is aligned
    __shared__ _Float16 Asmem[2][TILE_M * TILE_K];
    __shared__ _Float16 Bsmem[2][TILE_N * TILE_K];

    const int bz = blockIdx.z;
    const int bb = bz / nh, hh = bz % nh;
    const _Float16* A  = Abase + (long)bb * sAb + (long)hh * sAh;
    const _Float16* Bm = Bbase + (long)bb * sBb + (long)hh * sBh;
    float*    Cf = Cfbase ? Cfbase + (long)bb * sCb + (long)hh * sCh : nullptr;
    _Float16* Ch = Chbase ? Chbase + (long)bb * sCb + (long)hh * sCh : nullptr;

    const int m0 = blockIdx.y * TILE_M;
    const int n0 = blockIdx.x * TILE_N;

    const int tid  = threadIdx.x;
    const int wave = tid >> 5;
    const int lane = tid & 31;
    const int g    = lane >> 4;        // lane group (0: lanes0-15, 1: lanes16-31)
    const int l16  = lane & 15;
    const int wm   = (wave >> 1) * 32; // wave M offset inside block tile
    const int wn   = (wave & 1) * 64;  // wave N offset inside block tile

    // async ops issued per tile per thread: 2 (A) + 2 (B, BT only)
    constexpr int APT = BT ? 4 : 2;

    // stage one K-tile into buffer p (async for A and BT-B; sync transpose NN-B)
    auto stage = [&](int p, int k0) {
        unsigned ldsA = (unsigned)(size_t)(void*)&Asmem[p][0];
        unsigned ldsB = (unsigned)(size_t)(void*)&Bsmem[p][0];
        // A tile: 128 x 32 halves = 8KB = 512 x 16B chunks; M,K tile-multiples.
        for (int i = 0; i < 2; ++i) {
            int chunk = tid + i * 256;
            int r  = chunk >> 2;               // 4 chunks per row
            int c8 = (chunk & 3) << 3;
            async_copy16(ldsA + chunk * 16,
                         (const void*)(A + (long)(m0 + r) * lda + k0 + c8));
        }
        if constexpr (BT) {
            // B is (N,K): k-contiguous rows; BT call sites have N % 128 == 0.
            for (int i = 0; i < 2; ++i) {
                int chunk = tid + i * 256;
                int r  = chunk >> 2;
                int c8 = (chunk & 3) << 3;
                async_copy16(ldsB + chunk * 16,
                             (const void*)(Bm + (long)(n0 + r) * ldb + k0 + c8));
            }
        } else {
            // B is (K,N) (P@V): synchronous load + LDS transpose (N=64 ragged)
            for (int i = 0; i < 2; ++i) {
                int chunk = tid + i * 256;     // 512 chunks of 8 halves
                int kr = chunk >> 4;           // 16 chunks per k-row
                int n8 = (chunk & 15) << 3;
                int n  = n0 + n8;
                _Float16 tmp[8];
                if (n < N) {
                    v8h f = *reinterpret_cast<const v8h*>(Bm + (long)(k0 + kr) * ldb + n);
                    for (int j = 0; j < 8; ++j) tmp[j] = f[j];
                } else {
                    for (int j = 0; j < 8; ++j) tmp[j] = (_Float16)0.0f;
                }
                for (int j = 0; j < 8; ++j)
                    Bsmem[p][(n8 + j) * TILE_K + kr] = tmp[j];
            }
        }
    };

    v8f acc[2][4];
    for (int mi = 0; mi < 2; ++mi)
        for (int ni = 0; ni < 4; ++ni)
            for (int e = 0; e < 8; ++e) acc[mi][ni][e] = 0.0f;

    const int nt = K / TILE_K;
    stage(0, 0);                               // prologue: tile 0 in flight

    for (int kt = 0; kt < nt; ++kt) {
        const int p = kt & 1;
        // separate iter kt-1's readers of buf[1-p] from iter kt's writes to it
        if (kt > 0) __syncthreads();
        if (kt + 1 < nt) {
            stage(1 - p, (kt + 1) * TILE_K);   // tile k+1 in flight behind WMMAs
            WAIT_ASYNC(APT);                   // in-order: tile k has landed
        } else {
            WAIT_ASYNC(0);
        }
        __syncthreads();                       // all waves' tile-k copies visible

        // ---- fragments per CDNA5 VGPR layouts -> WMMA ----
        v16h afrag[2];
        for (int mi = 0; mi < 2; ++mi) {
            int r = wm + mi * 16 + l16;
            for (int v = 0; v < 8; ++v) {
                // 16-bit A 16x32: VGPR0-3 K=0..7 (+8 for lanes16-31),
                //                 VGPR4-7 K=16..23 (+8 for lanes16-31)
                int kk = (v < 4 ? 2 * v : 16 + 2 * (v - 4)) + g * 8;
                afrag[mi][2 * v]     = Asmem[p][r * TILE_K + kk];
                afrag[mi][2 * v + 1] = Asmem[p][r * TILE_K + kk + 1];
            }
        }
        for (int ni = 0; ni < 4; ++ni) {
            v16h bfrag;
            int c = wn + ni * 16 + l16;        // lanes hold N; lane group = K half
            for (int v = 0; v < 8; ++v) {
                int kk = (v < 4 ? 2 * v : 16 + 2 * (v - 4)) + g * 8;
                bfrag[2 * v]     = Bsmem[p][c * TILE_K + kk];
                bfrag[2 * v + 1] = Bsmem[p][c * TILE_K + kk + 1];
            }
            for (int mi = 0; mi < 2; ++mi)
                acc[mi][ni] = __builtin_amdgcn_wmma_f32_16x16x32_f16(
                    false, afrag[mi], false, bfrag, (short)0, acc[mi][ni], false, false);
        }
    }

    // ---- epilogue: alpha, bias, ReLU; C/D layout M = v + 8*g ----
    for (int ni = 0; ni < 4; ++ni) {
        int n = n0 + wn + ni * 16 + l16;
        if (n >= N) continue;
        float bv = bias ? bias[n] : 0.0f;
        for (int mi = 0; mi < 2; ++mi) {
            for (int v = 0; v < 8; ++v) {
                int m = m0 + wm + mi * 16 + g * 8 + v;
                float val = acc[mi][ni][v] * alpha + bv;
                if (RELU) val = fmaxf(val, 0.0f);
                if (Cf) Cf[(long)m * ldc + n] = val;
                if (Ch) Ch[(long)m * ldc + n] = (_Float16)val;
            }
        }
    }
    (void)M;
}

// ---------------------------------------------------------------------------
// streaming f32 -> f16 conversion (weights; once per launch)
// ---------------------------------------------------------------------------
__global__ void cvt_f16_kernel(const float* __restrict__ in,
                               _Float16* __restrict__ out, long n)
{
    long i = ((long)blockIdx.x * 256 + threadIdx.x) * 4;
    if (i + 3 < n) {
        float4 f = *reinterpret_cast<const float4*>(in + i);
        out[i + 0] = (_Float16)f.x;
        out[i + 1] = (_Float16)f.y;
        out[i + 2] = (_Float16)f.z;
        out[i + 3] = (_Float16)f.w;
    } else {
        for (; i < n; ++i) out[i] = (_Float16)in[i];
    }
}

// out[m,:] = emb[tok[m],:] * scale + pe[m % S,:]   (f32 master + f16 mirror)
__global__ void embed_kernel(const int* __restrict__ tok,
                             const float* __restrict__ emb,
                             const float* __restrict__ pe,
                             float* __restrict__ out32,
                             _Float16* __restrict__ out16,
                             int Dm, int Sm, float scale)
{
    long m = blockIdx.x;
    int  t = threadIdx.x;
    int  tk = tok[m];
    long s  = m % Sm;
    for (int i = 0; i < 2; ++i) {
        int d = t + i * 256;
        float v = emb[(long)tk * Dm + d] * scale + pe[s * Dm + d];
        out32[m * Dm + d] = v;
        out16[m * Dm + d] = (_Float16)v;
    }
}

// row-wise masked softmax over S=512 cols; reads f32 scores, writes f16 probs
__global__ void softmax_mask_kernel(const float* __restrict__ sc,
                                    _Float16* __restrict__ ph,
                                    const int* __restrict__ mask,
                                    int perq, int Sx, int nh)
{
    __shared__ float red[256];
    int q  = blockIdx.x;
    int bh = blockIdx.y;
    int b  = bh / nh;
    const float* row  = sc + ((long)bh * Sx + q) * Sx;
    _Float16*    prow = ph + ((long)bh * Sx + q) * Sx;
    const int* mrow = mask + (perq ? ((long)b * Sx + q) * Sx : (long)b * Sx);
    int t = threadIdx.x;

    float v0 = mrow[t]       ? row[t]       : -1e9f;
    float v1 = mrow[t + 256] ? row[t + 256] : -1e9f;

    red[t] = fmaxf(v0, v1);
    __syncthreads();
    for (int s = 128; s > 0; s >>= 1) { if (t < s) red[t] = fmaxf(red[t], red[t + s]); __syncthreads(); }
    float mx = red[0];
    __syncthreads();

    float e0 = __expf(v0 - mx), e1 = __expf(v1 - mx);
    red[t] = e0 + e1;
    __syncthreads();
    for (int s = 128; s > 0; s >>= 1) { if (t < s) red[t] += red[t + s]; __syncthreads(); }
    float inv = 1.0f / red[0];

    prow[t]       = (_Float16)(e0 * inv);
    prow[t + 256] = (_Float16)(e1 * inv);
}

// out = LayerNorm(x + r) * g + b  (D=512); writes f32 master + f16 mirror
__global__ void ln_residual_kernel(const float* __restrict__ x,
                                   const float* __restrict__ r,
                                   const float* __restrict__ g,
                                   const float* __restrict__ bta,
                                   float* __restrict__ out32,
                                   _Float16* __restrict__ out16)
{
    __shared__ float red[256];
    long m = blockIdx.x;
    int  t = threadIdx.x;
    float a0 = x[m * 512 + t]       + r[m * 512 + t];
    float a1 = x[m * 512 + t + 256] + r[m * 512 + t + 256];

    red[t] = a0 + a1;
    __syncthreads();
    for (int s = 128; s > 0; s >>= 1) { if (t < s) red[t] += red[t + s]; __syncthreads(); }
    float mean = red[0] * (1.0f / 512.0f);
    __syncthreads();

    float d0 = a0 - mean, d1 = a1 - mean;
    red[t] = d0 * d0 + d1 * d1;
    __syncthreads();
    for (int s = 128; s > 0; s >>= 1) { if (t < s) red[t] += red[t + s]; __syncthreads(); }
    float inv = rsqrtf(red[0] * (1.0f / 512.0f) + 1e-5f);

    float y0 = d0 * inv * g[t]       + bta[t];
    float y1 = d1 * inv * g[t + 256] + bta[t + 256];
    out32[m * 512 + t]       = y0;
    out32[m * 512 + t + 256] = y1;
    out16[m * 512 + t]       = (_Float16)y0;
    out16[m * 512 + t + 256] = (_Float16)y1;
}

// ---------------------------------------------------------------------------
// Host-side orchestration
// ---------------------------------------------------------------------------
static void launch_gemm(hipStream_t st,
                        const _Float16* A, int lda, long sAb, long sAh,
                        const _Float16* Bm, int ldb, long sBb, long sBh,
                        const float* bias,
                        float* Cf, _Float16* Ch, int ldc, long sCb, long sCh,
                        int M, int N, int K, float alpha,
                        int nh, int batch, bool bt, bool relu)
{
    dim3 grid((N + TILE_N - 1) / TILE_N, (M + TILE_M - 1) / TILE_M, batch);
    dim3 blk(256);
    if (bt) {
        if (relu)
            gemm_wmma_kernel<true, true><<<grid, blk, 0, st>>>(
                A, lda, sAb, sAh, Bm, ldb, sBb, sBh, bias, Cf, Ch, ldc, sCb, sCh, M, N, K, alpha, nh);
        else
            gemm_wmma_kernel<true, false><<<grid, blk, 0, st>>>(
                A, lda, sAb, sAh, Bm, ldb, sBb, sBh, bias, Cf, Ch, ldc, sCb, sCh, M, N, K, alpha, nh);
    } else {
        gemm_wmma_kernel<false, false><<<grid, blk, 0, st>>>(
            A, lda, sAb, sAh, Bm, ldb, sBb, sBh, bias, Cf, Ch, ldc, sCb, sCh, M, N, K, alpha, nh);
    }
}

extern "C" void kernel_launch(void* const* d_in, const int* in_sizes, int n_in,
                              void* d_out, int out_size, void* d_ws, size_t ws_size,
                              hipStream_t stream)
{
    (void)in_sizes; (void)n_in; (void)out_size; (void)ws_size;

    const int Dm = 512, NHh = 8, DKk = 64, DFFf = 2048, NEe = 6, NDd = 6, Vv = 32000, Bb = 4, Ss = 512;
    const int Mtok = Bb * Ss;                    // 2048 token rows
    const float scale = 22.62741699796952f;      // sqrt(512)

    auto F = [&](int i) { return (const float*)d_in[i]; };

    // flat input index map (insertion order of setup_inputs pytree):
    // 0 src_emb, 1 tgt_emb, 2 pe
    // enc L: base 3+16L: qw qb kw kb vw vb ow ob l1w l1b l2w l2b ln1g ln1b ln2g ln2b
    // dec L: base 99+26L: sa(8) ca(8) l1w l1b l2w l2b ln1g ln1b ln2g ln2b ln3g ln3b
    // 255 out.w, 256 out.b, 257 src, 258 tgt, 259 src_mask, 260 tgt_mask
    const float* src_emb = F(0);
    const float* tgt_emb = F(1);
    const float* pe      = F(2);
    const int* src      = (const int*)d_in[257];
    const int* tgt      = (const int*)d_in[258];
    const int* src_mask = (const int*)d_in[259];
    const int* tgt_mask = (const int*)d_in[260];

    // ---- bump allocator over d_ws (deterministic layout each call) ----
    char* cur = (char*)d_ws;
    auto alloc = [&](size_t bytes) -> void* {
        void* p = (void*)cur;
        cur += (bytes + 255) & ~(size_t)255;
        return p;
    };
    auto cvtW = [&](int idx, long n) -> const _Float16* {
        _Float16* h = (_Float16*)alloc((size_t)n * sizeof(_Float16));
        long q4 = (n + 3) / 4;
        int blocks = (int)((q4 + 255) / 256);
        cvt_f16_kernel<<<dim3(blocks), dim3(256), 0, stream>>>(F(idx), h, n);
        return h;
    };

    // ---- convert all GEMM weights to f16 once ----
    struct AttnW { const _Float16 *wq, *wk, *wv, *wo; const float *bq, *bk, *bv, *bo; };
    struct FfnW  { const _Float16 *l1, *l2; const float *b1, *b2; };

    AttnW encA[6]; FfnW encF[6];
    const float *encG1[6], *encB1[6], *encG2[6], *encB2[6];
    for (int L = 0; L < NEe; ++L) {
        int e = 3 + 16 * L;
        encA[L] = { cvtW(e + 0, (long)Dm * Dm), cvtW(e + 2, (long)Dm * Dm),
                    cvtW(e + 4, (long)Dm * Dm), cvtW(e + 6, (long)Dm * Dm),
                    F(e + 1), F(e + 3), F(e + 5), F(e + 7) };
        encF[L] = { cvtW(e + 8, (long)DFFf * Dm), cvtW(e + 10, (long)Dm * DFFf),
                    F(e + 9), F(e + 11) };
        encG1[L] = F(e + 12); encB1[L] = F(e + 13);
        encG2[L] = F(e + 14); encB2[L] = F(e + 15);
    }
    AttnW decSA[6], decCA[6]; FfnW decF[6];
    const float *decG1[6], *decB1[6], *decG2[6], *decB2[6], *decG3[6], *decB3[6];
    for (int L = 0; L < NDd; ++L) {
        int d = 99 + 26 * L;
        decSA[L] = { cvtW(d + 0, (long)Dm * Dm), cvtW(d + 2, (long)Dm * Dm),
                     cvtW(d + 4, (long)Dm * Dm), cvtW(d + 6, (long)Dm * Dm),
                     F(d + 1), F(d + 3), F(d + 5), F(d + 7) };
        decCA[L] = { cvtW(d + 8, (long)Dm * Dm), cvtW(d + 10, (long)Dm * Dm),
                     cvtW(d + 12, (long)Dm * Dm), cvtW(d + 14, (long)Dm * Dm),
                     F(d + 9), F(d + 11), F(d + 13), F(d + 15) };
        decF[L] = { cvtW(d + 16, (long)DFFf * Dm), cvtW(d + 18, (long)Dm * DFFf),
                    F(d + 17), F(d + 19) };
        decG1[L] = F(d + 20); decB1[L] = F(d + 21);
        decG2[L] = F(d + 22); decB2[L] = F(d + 23);
        decG3[L] = F(d + 24); decB3[L] = F(d + 25);
    }
    const _Float16* outw16 = cvtW(255, (long)Vv * Dm);
    const float*    outb   = F(256);

    // ---- activation buffers ----
    const long TOK = (long)Mtok * Dm;  // 1M elems
    float*    x32  = (float*)alloc(TOK * 4);             // encoder f32 master
    float*    y32  = (float*)alloc(TOK * 4);             // decoder f32 master
    float*    t2   = (float*)alloc(TOK * 4);             // branch output (f32)
    float*    sc32 = (float*)alloc((long)Bb * NHh * Ss * Ss * 4);  // scores f32
    _Float16* x16  = (_Float16*)alloc(TOK * 2);
    _Float16* y16  = (_Float16*)alloc(TOK * 2);
    _Float16* qh   = (_Float16*)alloc(TOK * 2);
    _Float16* kh   = (_Float16*)alloc(TOK * 2);
    _Float16* vh   = (_Float16*)alloc(TOK * 2);
    _Float16* t1h  = (_Float16*)alloc(TOK * 2);          // attention head-concat
    _Float16* Ph   = (_Float16*)alloc((long)Bb * NHh * Ss * Ss * 2); // probs f16
    _Float16* hidh = (_Float16*)alloc((long)Mtok * DFFf * 2);        // ffn hidden

    auto ln = [&](const float* x, const float* r, const float* g, const float* b,
                  float* o32, _Float16* o16) {
        ln_residual_kernel<<<dim3(Mtok), dim3(256), 0, stream>>>(x, r, g, b, o32, o16);
    };

    auto mha = [&](const _Float16* xq, const _Float16* xkv, const AttnW& Wt,
                   const int* mask, bool causal) {
        // Q/K/V projections -> f16 only
        launch_gemm(stream, xq,  Dm, 0, 0, Wt.wq, Dm, 0, 0, Wt.bq, nullptr, qh, Dm, 0, 0,
                    Mtok, Dm, Dm, 1.0f, 1, 1, true, false);
        launch_gemm(stream, xkv, Dm, 0, 0, Wt.wk, Dm, 0, 0, Wt.bk, nullptr, kh, Dm, 0, 0,
                    Mtok, Dm, Dm, 1.0f, 1, 1, true, false);
        launch_gemm(stream, xkv, Dm, 0, 0, Wt.wv, Dm, 0, 0, Wt.bv, nullptr, vh, Dm, 0, 0,
                    Mtok, Dm, Dm, 1.0f, 1, 1, true, false);
        // scores[b,h] = Q_h K_h^T / sqrt(DK): batched 512x512x64 -> f32
        launch_gemm(stream,
                    qh, Dm, (long)Ss * Dm, DKk,
                    kh, Dm, (long)Ss * Dm, DKk,
                    nullptr, sc32, nullptr, Ss, (long)NHh * Ss * Ss, (long)Ss * Ss,
                    Ss, Ss, DKk, 0.125f, NHh, Bb * NHh, true, false);
        softmax_mask_kernel<<<dim3(Ss, Bb * NHh), dim3(256), 0, stream>>>(
            sc32, Ph, mask, causal ? 1 : 0, Ss, NHh);
        // out_h = P V_h: batched 512x64x512 (NN), heads concatenated via ldc=D
        launch_gemm(stream,
                    Ph, Ss, (long)NHh * Ss * Ss, (long)Ss * Ss,
                    vh, Dm, (long)Ss * Dm, DKk,
                    nullptr, nullptr, t1h, Dm, (long)Ss * Dm, DKk,
                    Ss, DKk, Ss, 1.0f, NHh, Bb * NHh, false, false);
        // output projection -> f32 branch for LN
        launch_gemm(stream, t1h, Dm, 0, 0, Wt.wo, Dm, 0, 0, Wt.bo, t2, nullptr, Dm, 0, 0,
                    Mtok, Dm, Dm, 1.0f, 1, 1, true, false);
    };

    auto ffn = [&](const _Float16* x, const FfnW& Wt) {
        launch_gemm(stream, x, Dm, 0, 0, Wt.l1, Dm, 0, 0, Wt.b1, nullptr, hidh, DFFf, 0, 0,
                    Mtok, DFFf, Dm, 1.0f, 1, 1, true, true);      // ReLU
        launch_gemm(stream, hidh, DFFf, 0, 0, Wt.l2, DFFf, 0, 0, Wt.b2, t2, nullptr, Dm, 0, 0,
                    Mtok, Dm, DFFf, 1.0f, 1, 1, true, false);
    };

    // ---------------- encoder ----------------
    embed_kernel<<<dim3(Mtok), dim3(256), 0, stream>>>(src, src_emb, pe, x32, x16, Dm, Ss, scale);
    for (int L = 0; L < NEe; ++L) {
        mha(x16, x16, encA[L], src_mask, false);
        ln(x32, t2, encG1[L], encB1[L], x32, x16);
        ffn(x16, encF[L]);
        ln(x32, t2, encG2[L], encB2[L], x32, x16);
    }

    // ---------------- decoder ----------------
    embed_kernel<<<dim3(Mtok), dim3(256), 0, stream>>>(tgt, tgt_emb, pe, y32, y16, Dm, Ss, scale);
    for (int L = 0; L < NDd; ++L) {
        mha(y16, y16, decSA[L], tgt_mask, true);      // causal self-attn
        ln(y32, t2, decG1[L], decB1[L], y32, y16);
        mha(y16, x16, decCA[L], src_mask, false);     // cross-attn vs enc out
        ln(y32, t2, decG2[L], decB2[L], y32, y16);
        ffn(y16, decF[L]);
        ln(y32, t2, decG3[L], decB3[L], y32, y16);
    }

    // ---------------- output projection: (2048 x 32000 x 512) ----------------
    launch_gemm(stream, y16, Dm, 0, 0, outw16, Dm, 0, 0, outb,
                (float*)d_out, nullptr, Vv, 0, 0,
                Mtok, Vv, Dm, 1.0f, 1, 1, true, false);
}
